// MixtralMLP_25512105738342
// MI455X (gfx1250) — matrix operand order repaired and checked by reading, not compiled, not run on hardware
//
#include <hip/hip_runtime.h>
#include <hip/hip_bf16.h>

// ---------------------------------------------------------------------------
// Mixtral MoE MLP (top-2 of 8 experts), gather -> grouped GEMM -> scatter.
// T=8192 tokens, H=2048 hidden, I=4096 intermediate, E=8 experts, TOP_K=2.
// bf16 WMMA (v_wmma_f32_16x16x32_bf16) with on-the-fly fp32->bf16 conversion.
// Dual N-tile inner loops: 4 WMMAs per A-fragment in phase 1, 2 in phase 2.
// ---------------------------------------------------------------------------

#define T_ 8192
#define H_ 2048
#define I_ 4096
#define E_ 8
#define BM 16            // token tile per block (one WMMA M-tile)
#define NW 8             // waves per block (256 threads, wave32)
#define TILES_PER_E (T_ / BM)   // 512, worst case: all tokens on one expert
#define LDX (H_ + 8)     // padded LDS row stride (bf16 elems) for x tile
#define LDA (I_ + 8)     // padded LDS row stride (bf16 elems) for act tile

typedef float  v8f   __attribute__((ext_vector_type(8)));
typedef __bf16 v16bf __attribute__((ext_vector_type(16)));

// --- A-matrix fragment from LDS (bf16), contiguous K order matching B:
// lanes 0-15 hold row M=lane, K=0..15; lanes 16-31 same rows, K=16..31.
// VGPR v -> K=kb+2v,kb+2v+1
__device__ __forceinline__ v16bf load_a_lds(const __bf16* rowp, int k0, int kb) {
  v16bf a;
#pragma unroll
  for (int v = 0; v < 8; ++v) {
    int kk = k0 + kb + 2 * v;
    a[2 * v]     = rowp[kk];
    a[2 * v + 1] = rowp[kk + 1];
  }
  return a;
}

// --- B-matrix fragment from global fp32, converted to bf16.
// B 32x16 in contiguous K order (same as A): lanes 0-15 hold K=0..15 of column N=lane,
// lanes 16-31 hold K=16..31 of the same columns.  Caller passes pointer to
// W[row = ntile*16 + (lane&15)][k0 + (lane>=16 ? 16 : 0)] : 16 consecutive
// fp32 -> four global_load_b128 + eight v_cvt_pk_bf16_f32 per lane.
__device__ __forceinline__ v16bf load_b_g(const float* p) {
  v16bf b;
#pragma unroll
  for (int j = 0; j < 16; j += 4) {
    float4 f = *(const float4*)(p + j);
    b[j]     = (__bf16)f.x;
    b[j + 1] = (__bf16)f.y;
    b[j + 2] = (__bf16)f.z;
    b[j + 3] = (__bf16)f.w;
  }
  return b;
}

// --- SiLU(g)*u epilogue for one 16x16 C tile -> bf16 LDS.
// Fast path: v_exp_f32 + v_rcp_f32 (avoid IEEE v_div_scale ladder).
__device__ __forceinline__ void silu_mul_store(__bf16* act, int rowm, int col,
                                               int nt, v8f g, v8f u) {
#pragma unroll
  for (int r = 0; r < 8; ++r) {
    float gg = g[r], uu = u[r];
    float s = gg * __builtin_amdgcn_rcpf(1.f + __expf(-gg));
    act[(size_t)(rowm + r) * LDA + nt * 16 + col] = (__bf16)(s * uu);
  }
}

// ---------------------------------------------------------------------------
// Kernel 0: zero output (we accumulate with fp32 atomics) + expert counters.
// ---------------------------------------------------------------------------
__global__ void moe_zero_kernel(float* __restrict__ out, int* __restrict__ cnt) {
  size_t i = (size_t)blockIdx.x * blockDim.x + threadIdx.x;
  float4 z = {0.f, 0.f, 0.f, 0.f};
  ((float4*)out)[i] = z;          // grid sized exactly to T*H/4
  if (i < E_) cnt[i] = 0;
}

// ---------------------------------------------------------------------------
// Kernel 1: router — logits, top-2, renormalize, append to expert lists.
// One wave32 per token.
// ---------------------------------------------------------------------------
__global__ __launch_bounds__(256) void moe_router_kernel(
    const float* __restrict__ hs, const float* __restrict__ rw,
    int* __restrict__ cnt, int* __restrict__ tokList, float* __restrict__ sclList) {
  const int lane = threadIdx.x & 31;
  const int wv   = threadIdx.x >> 5;
  const int t    = blockIdx.x * NW + wv;

  float acc[E_];
#pragma unroll
  for (int e = 0; e < E_; ++e) acc[e] = 0.f;

  const float* x = hs + (size_t)t * H_;
  for (int h = lane * 4; h < H_; h += 32 * 4) {
    float4 xv = *(const float4*)(x + h);
#pragma unroll
    for (int e = 0; e < E_; ++e) {
      float4 w = *(const float4*)(rw + (size_t)e * H_ + h);
      acc[e] += xv.x * w.x + xv.y * w.y + xv.z * w.z + xv.w * w.w;
    }
  }
#pragma unroll
  for (int e = 0; e < E_; ++e)
#pragma unroll
    for (int off = 16; off > 0; off >>= 1)
      acc[e] += __shfl_xor(acc[e], off, 32);

  if (lane == 0) {
    int i1 = 0;
#pragma unroll
    for (int e = 1; e < E_; ++e) if (acc[e] > acc[i1]) i1 = e;
    int i2 = (i1 == 0) ? 1 : 0;
#pragma unroll
    for (int e = 0; e < E_; ++e) if (e != i1 && acc[e] > acc[i2]) i2 = e;
    // renormalized top-2 softmax == 2-way softmax of the two logits
    float w1 = 1.f / (1.f + __expf(acc[i2] - acc[i1]));
    float w2 = 1.f - w1;
    int p1 = atomicAdd(&cnt[i1], 1);
    tokList[i1 * T_ + p1] = t;  sclList[i1 * T_ + p1] = w1;
    int p2 = atomicAdd(&cnt[i2], 1);
    tokList[i2 * T_ + p2] = t;  sclList[i2 * T_ + p2] = w2;
  }
}

// ---------------------------------------------------------------------------
// Kernel 2: fused SwiGLU expert MLP over gathered token tiles.
// grid.x = E_ * TILES_PER_E ; block = 256 (8 waves).  ~197 KB dynamic LDS.
// ---------------------------------------------------------------------------
__global__ __launch_bounds__(256, 1) void moe_expert_kernel(
    const float* __restrict__ hs, const float* __restrict__ ws,
    const float* __restrict__ w2s, const int* __restrict__ cnt,
    const int* __restrict__ tokList, const float* __restrict__ sclList,
    float* __restrict__ out) {
  extern __shared__ char smem[];
  __bf16* xs  = (__bf16*)smem;                                   // [BM][LDX]
  __bf16* act = (__bf16*)(smem + (size_t)BM * LDX * 2);          // [BM][LDA]
  int*   stok = (int*)(smem + (size_t)BM * LDX * 2 + (size_t)BM * LDA * 2);
  float* sscl = (float*)(stok + BM);

  const int e    = blockIdx.x >> 9;          // / TILES_PER_E (512)
  const int tile = blockIdx.x & (TILES_PER_E - 1);
  const int c    = cnt[e];
  if (tile * BM >= c) return;                // empty tile for this expert

  const int tid  = threadIdx.x;
  const int lane = tid & 31;
  const int wv   = tid >> 5;

  if (tid < BM) {
    int m = tile * BM + tid;
    if (m < c) { stok[tid] = tokList[e * T_ + m]; sscl[tid] = sclList[e * T_ + m]; }
    else       { stok[tid] = 0;                   sscl[tid] = 0.f; }   // padded row
  }
  __syncthreads();

  // ---- gather: x[tokens] fp32 -> bf16 LDS tile ----
  for (int v = tid; v < BM * (H_ / 4); v += 256) {
    int row = v >> 9;                        // H_/4 = 512 float4 per row
    int c4  = (v & 511) * 4;
    float4 f = *(const float4*)(hs + (size_t)stok[row] * H_ + c4);
    __bf16* d = xs + (size_t)row * LDX + c4;
    d[0] = (__bf16)f.x; d[1] = (__bf16)f.y; d[2] = (__bf16)f.z; d[3] = (__bf16)f.w;
  }
  __syncthreads();

  const int col  = lane & 15;
  const int kb   = (lane >> 4) * 16;         // A K sub-base
  const int koff = (lane >> 4) * 16;         // B K sub-base
  const int rowm = (lane >> 4) * 8;          // C/D row base
  const __bf16* arow = xs + (size_t)(lane & 15) * LDX;

  // ---- phase 1: gate_up = x @ ws[e].T ; act = silu(gate)*up ----
  // Two nt-tiles (nt, nt+128) per pass: 4 WMMAs amortize each A-fragment.
  for (int nt = wv; nt < (I_ / 16) / 2; nt += NW) {
    const float* wg0 = ws + ((size_t)e * 2 * I_ + nt * 16 + col) * H_;
    const float* wu0 = wg0 + (size_t)I_ * H_;
    const float* wg1 = wg0 + (size_t)(I_ / 2) * H_;   // tile nt + 128
    const float* wu1 = wg1 + (size_t)I_ * H_;
    v8f ag0 = {}, au0 = {}, ag1 = {}, au1 = {};
    for (int k0 = 0; k0 < H_; k0 += 32) {
      v16bf a = load_a_lds(arow, k0, kb);
      __builtin_prefetch(wg0 + k0 + koff + 64, 0, 0);
      ag0 = __builtin_amdgcn_wmma_f32_16x16x32_bf16(false, a, false,
              load_b_g(wg0 + k0 + koff), (short)0, ag0, false, false);
      au0 = __builtin_amdgcn_wmma_f32_16x16x32_bf16(false, a, false,
              load_b_g(wu0 + k0 + koff), (short)0, au0, false, false);
      ag1 = __builtin_amdgcn_wmma_f32_16x16x32_bf16(false, a, false,
              load_b_g(wg1 + k0 + koff), (short)0, ag1, false, false);
      au1 = __builtin_amdgcn_wmma_f32_16x16x32_bf16(false, a, false,
              load_b_g(wu1 + k0 + koff), (short)0, au1, false, false);
    }
    silu_mul_store(act, rowm, col, nt,            ag0, au0);
    silu_mul_store(act, rowm, col, nt + I_ / 32,  ag1, au1);   // +128 tiles
  }
  __syncthreads();

  // ---- phase 2: y = act @ w2s[e].T ; scaled atomic scatter ----
  // Two ht-tiles (ht, ht+64) per pass: 2 WMMAs amortize each A-fragment.
  const __bf16* arow2 = act + (size_t)(lane & 15) * LDA;
  for (int ht = wv; ht < (H_ / 16) / 2; ht += NW) {
    const float* w2a = w2s + ((size_t)e * H_ + ht * 16 + col) * I_;
    const float* w2b = w2a + (size_t)(H_ / 2) * I_;   // tile ht + 64
    v8f acc0 = {}, acc1 = {};
    for (int k0 = 0; k0 < I_; k0 += 32) {
      v16bf a = load_a_lds(arow2, k0, kb);
      __builtin_prefetch(w2a + k0 + koff + 64, 0, 0);
      acc0 = __builtin_amdgcn_wmma_f32_16x16x32_bf16(false, a, false,
               load_b_g(w2a + k0 + koff), (short)0, acc0, false, false);
      acc1 = __builtin_amdgcn_wmma_f32_16x16x32_bf16(false, a, false,
               load_b_g(w2b + k0 + koff), (short)0, acc1, false, false);
    }
#pragma unroll
    for (int r = 0; r < 8; ++r) {
      int m = rowm + r;
      float v0 = acc0[r] * sscl[m];          // 0 for padded rows
      float v1 = acc1[r] * sscl[m];
      unsafeAtomicAdd(out + (size_t)stok[m] * H_ + ht * 16 + col, v0);
      unsafeAtomicAdd(out + (size_t)stok[m] * H_ + (ht + H_ / 32) * 16 + col, v1);
    }
  }
}

// ---------------------------------------------------------------------------
extern "C" void kernel_launch(void* const* d_in, const int* in_sizes, int n_in,
                              void* d_out, int out_size, void* d_ws, size_t ws_size,
                              hipStream_t stream) {
  const float* hs  = (const float*)d_in[0];   // [T,H]
  const float* rw  = (const float*)d_in[1];   // [E,H]
  const float* ws  = (const float*)d_in[2];   // [E,2I,H]
  const float* w2s = (const float*)d_in[3];   // [E,H,I]
  float* out = (float*)d_out;                 // [T,H]

  char* wsp = (char*)d_ws;
  int*   cnt = (int*)wsp;                                   // [E]
  int*   tok = (int*)(wsp + 256);                           // [E][T]
  float* scl = (float*)(wsp + 256 + (size_t)E_ * T_ * 4);   // [E][T]

  // zero output + counters
  moe_zero_kernel<<<(T_ * H_ / 4) / 256, 256, 0, stream>>>(out, cnt);

  // router: one wave per token
  moe_router_kernel<<<T_ / NW, 256, 0, stream>>>(hs, rw, cnt, tok, scl);

  // fused expert MLP
  size_t smem = (size_t)BM * LDX * 2 + (size_t)BM * LDA * 2 + BM * 4 + BM * 4;
  moe_expert_kernel<<<E_ * TILES_PER_E, 256, smem, stream>>>(
      hs, ws, w2s, cnt, tok, scl, out);
}
